// Seq2SeqModel_13322988552442
// MI455X (gfx1250) — matrix operand order repaired and checked by reading, not compiled, not run on hardware
//
#include <hip/hip_runtime.h>
#include <hip/hip_bf16.h>
#include <stdint.h>

// ---------------- problem dims ----------------
#define NV   256      // vocab
#define NE   128      // embed
#define NH   256      // hidden
#define NB   1024     // batch
#define NT   256      // encoder steps
#define NL   50       // decoder steps
#define KTOT 384      // E + H
#define NKT  12       // KTOT / 32
#define AROW_DW 200   // LDS row stride in dwords (192 data + 8 pad)

typedef __attribute__((ext_vector_type(16))) __bf16 v16bf;
typedef __attribute__((ext_vector_type(8)))  float  v8f;

union BFrag {
    uint4 q[2];
    v16bf v;
};

static __device__ __forceinline__ unsigned short f2bf(float f) {
    union { float f; uint32_t u; } v; v.f = f;
    uint32_t u = v.u;
    uint32_t r = (u + 0x7FFFu + ((u >> 16) & 1u)) >> 16;
    return (unsigned short)r;
}

static __device__ __forceinline__ uint32_t pack2bf(float lo, float hi) {
    return (uint32_t)f2bf(lo) | ((uint32_t)f2bf(hi) << 16);
}

static __device__ __forceinline__ float sigf(float x) {
    return 1.0f / (1.0f + __expf(-x));
}

// ---------------- prep: pack LSTM weights into WMMA-B fragment order ----------------
// B(k,n) = k < NE ? Wih[n*NE + k] : Whh[n*NH + (k-NE)]
// layout: dword d -> j=d&7 (K pair), lane=(d>>3)&31, nt=(d>>8)&63, kt=d>>14
// n = nt*16 + (lane&15); k = kt*32 + (lane>>4)*8 + (j&3)*2 + (j>>2)*16; low half = K even.
__global__ void pack_lstmB_kernel(const float* __restrict__ Wih,
                                  const float* __restrict__ Whh,
                                  uint32_t* __restrict__ out) {
    const int total = NKT * 64 * 32 * 8;   // 196608 dwords
    for (int d = blockIdx.x * blockDim.x + threadIdx.x; d < total;
         d += gridDim.x * blockDim.x) {
        int j    = d & 7;
        int lane = (d >> 3) & 31;
        int nt   = (d >> 8) & 63;
        int kt   = d >> 14;
        int n = nt * 16 + (lane & 15);
        int k = kt * 32 + (lane >> 4) * 8 + (j & 3) * 2 + (j >> 2) * 16;
        float v0 = (k     < NE) ? Wih[n * NE + k]     : Whh[n * NH + (k     - NE)];
        float v1 = (k + 1 < NE) ? Wih[n * NE + k + 1] : Whh[n * NH + (k + 1 - NE)];
        out[d] = pack2bf(v0, v1);
    }
}

// fc_W: [V x H]; K = H (8 ktiles), N = V (16 ntiles)
__global__ void pack_fcB_kernel(const float* __restrict__ W,
                                uint32_t* __restrict__ out) {
    const int total = 8 * 16 * 32 * 8;     // 32768 dwords
    for (int d = blockIdx.x * blockDim.x + threadIdx.x; d < total;
         d += gridDim.x * blockDim.x) {
        int j    = d & 7;
        int lane = (d >> 3) & 31;
        int nt   = (d >> 8) & 15;
        int kt   = d >> 12;
        int n = nt * 16 + (lane & 15);
        int k = kt * 32 + (lane >> 4) * 8 + (j & 3) * 2 + (j >> 2) * 16;
        out[d] = pack2bf(W[n * NH + k], W[n * NH + k + 1]);
    }
}

// embedding f32 -> bf16 row-major
__global__ void pack_emb_kernel(const float* __restrict__ emb,
                                uint32_t* __restrict__ out) {
    const int total = (NV * NE) / 2;       // 16384 dwords
    for (int d = blockIdx.x * blockDim.x + threadIdx.x; d < total;
         d += gridDim.x * blockDim.x) {
        out[d] = pack2bf(emb[2 * d], emb[2 * d + 1]);
    }
}

// zero h0 (bf16), zero c (f32), seed decoder token = input_seq[:,0]
__global__ void init_state_kernel(uint32_t* __restrict__ h0,
                                  float* __restrict__ c,
                                  int* __restrict__ tok,
                                  const int* __restrict__ seq) {
    const int stride = gridDim.x * blockDim.x;
    int i = blockIdx.x * blockDim.x + threadIdx.x;
    for (int d = i; d < (NB * NH) / 2; d += stride) h0[d] = 0u;
    for (int d = i; d < NB * NH; d += stride)       c[d]  = 0.0f;
    for (int d = i; d < NB; d += stride)            tok[d] = seq[d * NT];
}

// ---------------- fused LSTM step ----------------
// gates = [x_t | h] @ [Wih;Whh]^T + b ; cell update; h ping-pong, c in place.
// grid = 32 blocks x 256 threads. blockIdx: mblk = bx>>1 (64 rows), hhalf = bx&1.
// wave w handles h-columns h0 = hhalf*128 + w*16 across all 4 gates.
__global__ __launch_bounds__(256) void lstm_step_kernel(
    const int* __restrict__ tok_src, int tok_stride, int t, int tok_mode,
    const unsigned short* __restrict__ embB,   // V x E bf16
    const uint32_t* __restrict__ WB,           // packed B frags [kt][nt=64][lane][8]
    const float* __restrict__ bias,            // 4H
    const unsigned short* __restrict__ h_in,   // B x H bf16
    unsigned short* __restrict__ h_out,        // B x H bf16
    float* __restrict__ c)                     // B x H f32
{
    __shared__ uint32_t AP[64 * AROW_DW];
    __shared__ int toks[64];

    const int tid = threadIdx.x;
    const int m0  = (blockIdx.x >> 1) * 64;
    const int hhalf = blockIdx.x & 1;

    if (tid < 64) {
        int b = m0 + tid;
        toks[tid] = tok_mode ? tok_src[b] : tok_src[b * tok_stride + t];
    }
    __syncthreads();

    // Stage A panel: row r = [ emb[tok[r]] (64 dw) | h_in[m0+r] (128 dw) ]
    for (int idx = tid; idx < 64 * 192; idx += 256) {
        int r = idx / 192;
        int cp = idx - r * 192;
        uint32_t v;
        if (cp < 64) {
            v = ((const uint32_t*)(embB + toks[r] * NE))[cp];
        } else {
            v = ((const uint32_t*)(h_in + (m0 + r) * NH))[cp - 64];
        }
        AP[r * AROW_DW + cp] = v;
    }
    __syncthreads();

    const int wave = tid >> 5;
    const int lane = tid & 31;
    const int lrow = lane & 15;
    const int lhi  = lane >> 4;
    const int h0   = hhalf * 128 + wave * 16;
    const int ntbase = h0 >> 4;        // n-tile within each gate block of 16 tiles

    v8f acc[4][4];                     // [gate][mtile]
    #pragma unroll
    for (int g = 0; g < 4; ++g)
        #pragma unroll
        for (int mt = 0; mt < 4; ++mt)
            acc[g][mt] = (v8f){0.f,0.f,0.f,0.f,0.f,0.f,0.f,0.f};

    for (int kt = 0; kt < NKT; ++kt) {
        // A fragments from LDS: lanes 0-15 K chunk {0..7,16..23}; lanes 16-31 {8..15,24..31}
        v16bf a[4];
        #pragma unroll
        for (int mt = 0; mt < 4; ++mt) {
            const uint32_t* p = &AP[(mt * 16 + lrow) * AROW_DW + kt * 16 + lhi * 4];
            BFrag f;
            f.q[0] = *(const uint4*)(p);       // K .. K+7
            f.q[1] = *(const uint4*)(p + 8);   // K+16 .. K+23
            a[mt] = f.v;
        }
        // B fragments (one per gate) from packed weights
        v16bf bfr[4];
        #pragma unroll
        for (int g = 0; g < 4; ++g) {
            int nt = ntbase + g * 16;
            const uint32_t* p = WB + (((size_t)kt * 64 + nt) * 32 + lane) * 8;
            BFrag f;
            f.q[0] = *(const uint4*)(p);
            f.q[1] = *(const uint4*)(p + 4);
            bfr[g] = f.v;
            if (kt + 1 < NKT)
                __builtin_prefetch(WB + ((((size_t)kt + 1) * 64 + nt) * 32 + lane) * 8, 0, 0);
        }
        #pragma unroll
        for (int g = 0; g < 4; ++g)
            #pragma unroll
            for (int mt = 0; mt < 4; ++mt)
                acc[g][mt] = __builtin_amdgcn_wmma_f32_16x16x32_bf16(
                    false, a[mt], false, bfr[g], (short)0, acc[g][mt], false, false);
    }

    // fused cell update; accumulator element (r, lane) -> m = m0+mt*16+lhi*8+r, n = h0+lrow
    float bi[4];
    #pragma unroll
    for (int g = 0; g < 4; ++g) bi[g] = bias[g * NH + h0 + lrow];
    const int ncol = h0 + lrow;
    #pragma unroll
    for (int mt = 0; mt < 4; ++mt) {
        #pragma unroll
        for (int r = 0; r < 8; ++r) {
            int m = m0 + mt * 16 + lhi * 8 + r;
            float iv = sigf(acc[0][mt][r] + bi[0]);
            float fv = sigf(acc[1][mt][r] + bi[1]);
            float gv = tanhf(acc[2][mt][r] + bi[2]);
            float ov = sigf(acc[3][mt][r] + bi[3]);
            float cold = c[m * NH + ncol];
            float cnew = fv * cold + iv * gv;
            c[m * NH + ncol] = cnew;
            h_out[m * NH + ncol] = f2bf(ov * tanhf(cnew));
        }
    }
}

// ---------------- decoder logits: h @ fc_W^T + fc_b ----------------
// 64 waves; wave gid: m0 = (gid>>2)*64, n0 = (gid&3)*64. grid = 8 x 256.
__global__ __launch_bounds__(256) void logits_kernel(
    const unsigned short* __restrict__ h,   // B x H bf16
    const uint32_t* __restrict__ fcB,       // packed [kt=8][nt=16][lane][8]
    const float* __restrict__ fc_b,
    float* __restrict__ out, int step)
{
    const int tid  = threadIdx.x;
    const int wave = tid >> 5;
    const int lane = tid & 31;
    const int lrow = lane & 15;
    const int lhi  = lane >> 4;
    const int gid  = blockIdx.x * 8 + wave;
    const int m0   = (gid >> 2) * 64;
    const int n0   = (gid & 3) * 64;

    v8f acc[4][4];                       // [ntile][mtile]
    #pragma unroll
    for (int nt = 0; nt < 4; ++nt)
        #pragma unroll
        for (int mt = 0; mt < 4; ++mt)
            acc[nt][mt] = (v8f){0.f,0.f,0.f,0.f,0.f,0.f,0.f,0.f};

    for (int kt = 0; kt < 8; ++kt) {
        v16bf a[4];
        #pragma unroll
        for (int mt = 0; mt < 4; ++mt) {
            int m = m0 + mt * 16 + lrow;
            const uint32_t* p = (const uint32_t*)(h + m * NH + kt * 32 + lhi * 8);
            BFrag f;
            f.q[0] = *(const uint4*)(p);
            f.q[1] = *(const uint4*)(p + 8);
            a[mt] = f.v;
        }
        v16bf bfr[4];
        #pragma unroll
        for (int nt = 0; nt < 4; ++nt) {
            int ntg = (n0 >> 4) + nt;
            const uint32_t* p = fcB + (((size_t)kt * 16 + ntg) * 32 + lane) * 8;
            BFrag f;
            f.q[0] = *(const uint4*)(p);
            f.q[1] = *(const uint4*)(p + 4);
            bfr[nt] = f.v;
        }
        #pragma unroll
        for (int nt = 0; nt < 4; ++nt)
            #pragma unroll
            for (int mt = 0; mt < 4; ++mt)
                acc[nt][mt] = __builtin_amdgcn_wmma_f32_16x16x32_bf16(
                    false, a[mt], false, bfr[nt], (short)0, acc[nt][mt], false, false);
    }

    #pragma unroll
    for (int nt = 0; nt < 4; ++nt) {
        int n = n0 + nt * 16 + lrow;
        float bv = fc_b[n];
        #pragma unroll
        for (int mt = 0; mt < 4; ++mt)
            #pragma unroll
            for (int r = 0; r < 8; ++r) {
                int m = m0 + mt * 16 + lhi * 8 + r;
                out[(size_t)m * (NL * NV) + step * NV + n] = acc[nt][mt][r] + bv;
            }
    }
}

// ---------------- argmax (first max, like jnp.argmax) ----------------
__global__ void argmax_kernel(const float* __restrict__ out, int step,
                              int* __restrict__ tok) {
    int b = blockIdx.x * blockDim.x + threadIdx.x;
    if (b >= NB) return;
    const float* row = out + (size_t)b * (NL * NV) + step * NV;
    float best = row[0];
    int bi = 0;
    for (int v = 1; v < NV; ++v) {
        float x = row[v];
        if (x > best) { best = x; bi = v; }
    }
    tok[b] = bi;
}

// ---------------- host ----------------
extern "C" void kernel_launch(void* const* d_in, const int* in_sizes, int n_in,
                              void* d_out, int out_size, void* d_ws, size_t ws_size,
                              hipStream_t stream) {
    (void)in_sizes; (void)n_in; (void)out_size; (void)ws_size;
    const int*   seq  = (const int*)d_in[0];
    const float* emb  = (const float*)d_in[1];
    const float* eWih = (const float*)d_in[2];
    const float* eWhh = (const float*)d_in[3];
    const float* eb   = (const float*)d_in[4];
    const float* dWih = (const float*)d_in[5];
    const float* dWhh = (const float*)d_in[6];
    const float* db   = (const float*)d_in[7];
    const float* fcW  = (const float*)d_in[8];
    const float* fcb  = (const float*)d_in[9];
    float* out = (float*)d_out;

    char* ws = (char*)d_ws;
    size_t o = 0;
    auto take = [&](size_t bytes) { char* p = ws + o; o += (bytes + 255) & ~(size_t)255; return p; };
    uint32_t*       encB = (uint32_t*)      take(NKT * 64 * 32 * 8 * 4);    // 768 KB
    uint32_t*       decB = (uint32_t*)      take(NKT * 64 * 32 * 8 * 4);    // 768 KB
    uint32_t*       fcB  = (uint32_t*)      take(8 * 16 * 32 * 8 * 4);      // 128 KB
    unsigned short* embB = (unsigned short*)take(NV * NE * 2);              //  64 KB
    unsigned short* hA   = (unsigned short*)take(NB * NH * 2);              // 512 KB
    unsigned short* hB   = (unsigned short*)take(NB * NH * 2);              // 512 KB
    float*          cst  = (float*)         take(NB * NH * 4);              //   1 MB
    int*            tok  = (int*)           take(NB * 4);

    // prep
    pack_lstmB_kernel<<<256, 256, 0, stream>>>(eWih, eWhh, encB);
    pack_lstmB_kernel<<<256, 256, 0, stream>>>(dWih, dWhh, decB);
    pack_fcB_kernel  <<<64, 256, 0, stream>>>(fcW, fcB);
    pack_emb_kernel  <<<64, 256, 0, stream>>>(emb, (uint32_t*)embB);
    init_state_kernel<<<256, 256, 0, stream>>>((uint32_t*)hA, cst, tok, seq);

    // encoder
    const unsigned short* hin = hA;
    unsigned short*       hout = hB;
    for (int t = 0; t < NT; ++t) {
        lstm_step_kernel<<<32, 256, 0, stream>>>(seq, NT, t, 0, embB, encB, eb,
                                                 hin, hout, cst);
        const unsigned short* tmp = hout; hout = (unsigned short*)hin; hin = tmp;
    }

    // decoder
    for (int s = 0; s < NL; ++s) {
        lstm_step_kernel<<<32, 256, 0, stream>>>(tok, 0, 0, 1, embB, decB, db,
                                                 hin, hout, cst);
        const unsigned short* tmp = hout; hout = (unsigned short*)hin; hin = tmp;
        logits_kernel<<<8, 256, 0, stream>>>(hin, fcB, fcb, out, s);
        argmax_kernel<<<4, 256, 0, stream>>>(out, s, tok);
    }
}
